// Attention_15393162789187
// MI455X (gfx1250) — matrix operand order repaired and checked by reading, not compiled, not run on hardware
//
#include <hip/hip_runtime.h>
#include <hip/hip_bf16.h>

typedef __attribute__((ext_vector_type(16))) __bf16 v16bf;
typedef __attribute__((ext_vector_type(8)))  __bf16 v8bf;
typedef __attribute__((ext_vector_type(8)))  float  v8f;

constexpr int Bc  = 4;
constexpr int Sc  = 2048;
constexpr int Dc  = 1024;
constexpr int Hc  = 16;
constexpr int DKc = 64;

// ---------------- WMMA helpers (CDNA5 16x16x32 bf16, f32 accum) ----------------

__device__ inline v8f wmma_bf16(v16bf a, v16bf b, v8f c) {
  return __builtin_amdgcn_wmma_f32_16x16x32_bf16(
      /*neg_a=*/false, a, /*neg_b=*/false, b,
      /*c_mod=*/(short)0, c, /*reuse_a=*/false, /*reuse_b=*/false);
}

// A-matrix 16x32 bf16 fragment (contiguous k order, same as B): lane L holds row m=L&15;
// element e -> k = hl*16 + e
__device__ inline v16bf load_frag_a(const __bf16* base, int ldk, int lane) {
  const int m  = lane & 15;
  const int hl = lane >> 4;
  const __bf16* p = base + (size_t)m * ldk + hl * 16;
  union { v16bf v; v8bf h[2]; } u;
  u.h[0] = *(const v8bf*)(p);
  u.h[1] = *(const v8bf*)(p + 8);
  return u.v;
}

// B-matrix 32x16 bf16 fragment: lane L holds col n=L&15; element e -> k = hl*16 + e
// (contiguous 16 bf16 along K in the row-major [N][K] source)
__device__ inline v16bf load_frag_b(const __bf16* base, int ldk, int lane) {
  const int n  = lane & 15;
  const int hl = lane >> 4;
  return *(const v16bf*)(base + (size_t)n * ldk + hl * 16);
}

__device__ inline float rowmax16(float v) {
  #pragma unroll
  for (int m = 1; m < 16; m <<= 1) v = fmaxf(v, __shfl_xor(v, m, 32));
  return v;
}

// ---------------- f32 -> bf16 conversion ----------------

__global__ __launch_bounds__(256) void cvt_bf16_kernel(const float* __restrict__ in,
                                                       __bf16* __restrict__ out, int n) {
  int i = blockIdx.x * 256 + threadIdx.x;
  if (i < n) out[i] = (__bf16)in[i];
}

// ---------------- GEMM: Y = A (MxK) * W^T  (W is [N][K] row major) ----------------
// Wave tile: 32(M) x 64(N): 2 A-frags + 4 B-frags -> 8 WMMAs per 32-K step.
// MODE 0: store bf16 as Q/K layout [bh][s][dk]
// MODE 1: store bf16 as V-transposed layout [bh][dk][s]
// MODE 2: store f32 row-major [M][N] (final output)
template<int MODE>
__global__ __launch_bounds__(128) void gemm32_kernel(const __bf16* __restrict__ A,
                                                     const __bf16* __restrict__ W,
                                                     void* __restrict__ outp) {
  const int lane = threadIdx.x & 31;
  const int wid  = threadIdx.x >> 5;
  const int gw   = blockIdx.x * 4 + wid;          // global wave id
  const int TN   = Dc / 64;                        // 16 n-strips of 64
  const int m0   = (gw / TN) * 32;
  const int n0   = (gw % TN) * 64;

  v8f acc[2][4];
  #pragma unroll
  for (int i = 0; i < 2; ++i)
    #pragma unroll
    for (int j = 0; j < 4; ++j) acc[i][j] = v8f{0};

  const __bf16* arow0 = A + (size_t)m0 * Dc;
  const __bf16* arow1 = A + (size_t)(m0 + 16) * Dc;

  for (int kk = 0; kk < Dc; kk += 32) {
    // prefetch the streaming A operand a few steps ahead (global_prefetch_b8)
    if (kk + 256 < Dc) {
      __builtin_prefetch(arow0 + kk + 256, 0, 3);
      __builtin_prefetch(arow1 + kk + 256, 0, 3);
    }
    v16bf a0 = load_frag_a(arow0 + kk, Dc, lane);
    v16bf a1 = load_frag_a(arow1 + kk, Dc, lane);
    #pragma unroll
    for (int j = 0; j < 4; ++j) {
      v16bf b = load_frag_b(W + (size_t)(n0 + j * 16) * Dc + kk, Dc, lane);
      acc[0][j] = wmma_bf16(a0, b, acc[0][j]);
      acc[1][j] = wmma_bf16(a1, b, acc[1][j]);
    }
  }

  const int nl = lane & 15;
  const int hl = lane >> 4;
  #pragma unroll
  for (int i = 0; i < 2; ++i) {
    #pragma unroll
    for (int j = 0; j < 4; ++j) {
      #pragma unroll
      for (int r = 0; r < 8; ++r) {
        const int m = m0 + i * 16 + r + 8 * hl;
        const int n = n0 + j * 16 + nl;
        const float v = acc[i][j][r];
        if (MODE == 2) {
          ((float*)outp)[(size_t)m * Dc + n] = v;
        } else {
          const int b  = m >> 11;            // /Sc
          const int s  = m & (Sc - 1);
          const int h  = n >> 6;             // /DKc
          const int dk = n & (DKc - 1);
          size_t idx;
          if (MODE == 0) idx = (((size_t)b * Hc + h) * Sc + s) * DKc + dk;   // [bh][s][dk]
          else           idx = (((size_t)b * Hc + h) * DKc + dk) * Sc + s;   // [bh][dk][s]
          ((__bf16*)outp)[idx] = (__bf16)v;
        }
      }
    }
  }
}

// ---------------- RoPE (in place on [bh][s][dk] bf16, interleaved pairs) ----------------

__global__ __launch_bounds__(256) void rope_kernel(__bf16* __restrict__ T,
                                                   const int* __restrict__ pos) {
  const int i = blockIdx.x * 256 + threadIdx.x;    // i = (bh*Sc + s)*32 + pair
  const int p = i & 31;
  const int s = (i >> 5) & (Sc - 1);
  const size_t base = (size_t)(i >> 5) * DKc + 2 * p;
  const float inv_freq = __powf(10000.0f, -(float)(2 * p) / (float)DKc);
  const float ang = (float)pos[s] * inv_freq;
  float sn, cs;
  __sincosf(ang, &sn, &cs);
  const float x1 = (float)T[base];
  const float x2 = (float)T[base + 1];
  T[base]     = (__bf16)(x1 * cs - x2 * sn);
  T[base + 1] = (__bf16)(x2 * cs + x1 * sn);
}

// ---------------- Flash attention step: one 32-key block for a 16-query tile ----------------
// o[0..3] accumulate P*V (16x64); o[4] accumulates P*ones = running row-sum of P
// (the softmax denominator), so no cross-lane row-sum reduction is needed.

template<bool MASKED>
__device__ inline void attn_step(int kb, int q0, int lane, int nl, int hl,
                                 const __bf16* __restrict__ Kbh,   // K base of this (b,h): [s][dk]
                                 const __bf16* __restrict__ Vbh,   // Vt base of this (b,h): [dk][s]
                                 v16bf aq0, v16bf aq1, v16bf ones, __bf16* pl,
                                 v8f (&o)[5], float (&rmax)[8]) {
  // ---- scores: 16 queries x 32 keys, K-dim = DK = 64 (2 chunks of 32) ----
  const __bf16* kptr = Kbh + (size_t)kb * DKc;
  v8f s0 = v8f{0}, s1 = v8f{0};
  s0 = wmma_bf16(aq0, load_frag_b(kptr,                 DKc, lane), s0);
  s0 = wmma_bf16(aq1, load_frag_b(kptr + 32,            DKc, lane), s0);
  s1 = wmma_bf16(aq0, load_frag_b(kptr + 16 * DKc,      DKc, lane), s1);
  s1 = wmma_bf16(aq1, load_frag_b(kptr + 16 * DKc + 32, DKc, lane), s1);

  // ---- (optional) causal mask + scale + online softmax max-tracking ----
  const int k0c = kb + nl;
  const int k1c = kb + 16 + nl;
  #pragma unroll
  for (int r = 0; r < 8; ++r) {
    float v0, v1;
    if (MASKED) {
      const int qr = q0 + r + 8 * hl;
      v0 = (k0c <= qr) ? s0[r] * 0.125f : -__builtin_inff();
      v1 = (k1c <= qr) ? s1[r] * 0.125f : -__builtin_inff();
    } else {
      v0 = s0[r] * 0.125f;
      v1 = s1[r] * 0.125f;
    }
    const float mx = rowmax16(fmaxf(v0, v1));
    const float mnew = fmaxf(rmax[r], mx);
    const float sc = __expf(rmax[r] - mnew);     // first iter: exp(-inf)=0
    rmax[r] = mnew;
    const float p0 = __expf(v0 - mnew);
    const float p1 = __expf(v1 - mnew);
    o[0][r] *= sc; o[1][r] *= sc; o[2][r] *= sc; o[3][r] *= sc; o[4][r] *= sc;
    pl[(r + 8 * hl) * 32 + nl]      = (__bf16)p0;
    pl[(r + 8 * hl) * 32 + 16 + nl] = (__bf16)p1;
  }

  // intra-wave LDS write -> read ordering
  asm volatile("s_wait_dscnt 0" ::: "memory");

  // ---- O += P (16x32) * V (32x64), denom += P * ones ----
  const v16bf pa = load_frag_a(pl, 32, lane);
  const __bf16* vbase = Vbh + kb;
  #pragma unroll
  for (int j = 0; j < 4; ++j)
    o[j] = wmma_bf16(pa, load_frag_b(vbase + (size_t)(j * 16) * Sc, Sc, lane), o[j]);
  o[4] = wmma_bf16(pa, ones, o[4]);              // row-sum fused into the matrix pipe
}

// ---------------- Flash attention: one wave = one 16-query tile of one (b,h) ----------------

__global__ __launch_bounds__(256) void attn_kernel(const __bf16* __restrict__ Qb,
                                                   const __bf16* __restrict__ Kb,
                                                   const __bf16* __restrict__ Vt,
                                                   __bf16* __restrict__ Ob) {
  __shared__ __bf16 plds[8 * 16 * 32];             // per-wave 16x32 P staging
  const int lane = threadIdx.x & 31;
  const int wid  = threadIdx.x >> 5;
  const int gw   = blockIdx.x * 8 + wid;
  const int bh   = gw >> 7;                        // Sc/16 = 128 q-tiles per (b,h)
  const int q0   = (gw & 127) * 16;
  const int nl   = lane & 15;
  const int hl   = lane >> 4;

  const __bf16* Kbh = Kb + (size_t)bh * Sc * DKc;
  const __bf16* Vbh = Vt + (size_t)bh * DKc * Sc;
  const __bf16* qrow = Qb + ((size_t)bh * Sc + q0) * DKc;
  const v16bf aq0 = load_frag_a(qrow,      DKc, lane);   // dk 0..31
  const v16bf aq1 = load_frag_a(qrow + 32, DKc, lane);   // dk 32..63

  v16bf ones;
  #pragma unroll
  for (int e = 0; e < 16; ++e) ones[e] = (__bf16)1.0f;

  v8f o[5] = {v8f{0}, v8f{0}, v8f{0}, v8f{0}, v8f{0}};
  float rmax[8];
  #pragma unroll
  for (int r = 0; r < 8; ++r) rmax[r] = -__builtin_inff();

  __bf16* pl = plds + wid * (16 * 32);

  // key blocks with max key (kb+31) <= min query (q0) need no causal mask
  const int kfull = q0 & ~31;
  for (int kb = 0; kb < kfull; kb += 32)
    attn_step<false>(kb, q0, lane, nl, hl, Kbh, Vbh, aq0, aq1, ones, pl, o, rmax);
  for (int kb = kfull; kb < q0 + 16; kb += 32)
    attn_step<true >(kb, q0, lane, nl, hl, Kbh, Vbh, aq0, aq1, ones, pl, o, rmax);

  // ---- normalize by the fused denominator + store O as [b][s][d] in bf16 ----
  const int b = bh >> 4;
  const int h = bh & 15;
  #pragma unroll
  for (int r = 0; r < 8; ++r) {
    const float inv = 1.0f / o[4][r];              // every lane holds its row's sum
    const int s = q0 + r + 8 * hl;
    #pragma unroll
    for (int j = 0; j < 4; ++j) {
      const int d = h * DKc + j * 16 + nl;
      Ob[((size_t)b * Sc + s) * Dc + d] = (__bf16)(o[j][r] * inv);
    }
  }
}

// ---------------- launch ----------------

extern "C" void kernel_launch(void* const* d_in, const int* in_sizes, int n_in,
                              void* d_out, int out_size, void* d_ws, size_t ws_size,
                              hipStream_t stream) {
  const float* x   = (const float*)d_in[0];
  const int*   pos = (const int*)d_in[1];
  const float* Wq  = (const float*)d_in[2];
  const float* Wk  = (const float*)d_in[3];
  const float* Wv  = (const float*)d_in[4];
  const float* Wo  = (const float*)d_in[5];

  char* ws = (char*)d_ws;
  const size_t MB = 1024 * 1024;
  __bf16* xb  = (__bf16*)(ws + 0 * MB);    // 16 MB  (B*S*D bf16)
  __bf16* wqb = (__bf16*)(ws + 16 * MB);   // 2 MB
  __bf16* wkb = (__bf16*)(ws + 18 * MB);
  __bf16* wvb = (__bf16*)(ws + 20 * MB);
  __bf16* wob = (__bf16*)(ws + 22 * MB);
  __bf16* Qb  = (__bf16*)(ws + 24 * MB);   // 16 MB, [bh][s][dk]
  __bf16* Kb  = (__bf16*)(ws + 40 * MB);   // 16 MB, [bh][s][dk]
  __bf16* Vt  = (__bf16*)(ws + 56 * MB);   // 16 MB, [bh][dk][s]
  __bf16* Ob  = (__bf16*)(ws + 72 * MB);   // 16 MB, [b][s][d]

  const int nx = Bc * Sc * Dc;             // 8388608
  const int nw = Dc * Dc;                  // 1048576
  cvt_bf16_kernel<<<nx / 256, 256, 0, stream>>>(x,  xb,  nx);
  cvt_bf16_kernel<<<nw / 256, 256, 0, stream>>>(Wq, wqb, nw);
  cvt_bf16_kernel<<<nw / 256, 256, 0, stream>>>(Wk, wkb, nw);
  cvt_bf16_kernel<<<nw / 256, 256, 0, stream>>>(Wv, wvb, nw);
  cvt_bf16_kernel<<<nw / 256, 256, 0, stream>>>(Wo, wob, nw);

  const int gemm_blocks = ((Bc * Sc / 32) * (Dc / 64)) / 4;   // 1024 blocks of 4 waves
  gemm32_kernel<0><<<gemm_blocks, 128, 0, stream>>>(xb, wqb, (void*)Qb);
  gemm32_kernel<0><<<gemm_blocks, 128, 0, stream>>>(xb, wkb, (void*)Kb);
  gemm32_kernel<1><<<gemm_blocks, 128, 0, stream>>>(xb, wvb, (void*)Vt);

  const int nrope = Bc * Hc * Sc * (DKc / 2);  // 4194304 pair-threads
  rope_kernel<<<nrope / 256, 256, 0, stream>>>(Qb, pos);
  rope_kernel<<<nrope / 256, 256, 0, stream>>>(Kb, pos);

  const int attn_blocks = (Bc * Hc * (Sc / 16)) / 8;          // 1024 blocks of 8 waves
  attn_kernel<<<attn_blocks, 256, 0, stream>>>(Qb, Kb, Vt, Ob);

  gemm32_kernel<2><<<gemm_blocks, 128, 0, stream>>>(Ob, wob, d_out);
}
